// SwinTransformer_4793183502357
// MI455X (gfx1250) — compile-verified, matrix operand order checked by reading, not run on hardware
//
#include <hip/hip_runtime.h>
#include <math.h>

typedef _Float16 h16;
typedef __attribute__((ext_vector_type(16))) _Float16 v16h;
typedef __attribute__((ext_vector_type(8)))  _Float16 v8h;
typedef __attribute__((ext_vector_type(4)))  _Float16 v4h;
typedef __attribute__((ext_vector_type(8)))  float    v8f;

#define EPI_BIAS       0
#define EPI_BIAS_GELU  1
#define EPI_BIAS_RES   2

// CDNA5 async global->LDS copy path (ASYNCcnt), with synchronous fallback.
#if defined(__gfx1250__) && __has_builtin(__builtin_amdgcn_global_load_async_to_lds_b128) && __has_builtin(__builtin_amdgcn_s_wait_asynccnt)
#define SWIN_ASYNC 1
#else
#define SWIN_ASYNC 0
#endif

#if SWIN_ASYNC
// Builtin signature (from hipcc diagnostic): first param is
//   __attribute__((vector_size(16))) int __device__ *   (AS1, int4)
// second is the LDS-side int4 pointer (AS3), then imm offset / imm cpol.
typedef int swin_v4i __attribute__((vector_size(16)));
typedef __attribute__((address_space(1))) swin_v4i* swin_g128p;
typedef __attribute__((address_space(3))) swin_v4i* swin_l128p;
#endif

__device__ __forceinline__ void swin_cp16(const h16* g, h16* l) {
#if SWIN_ASYNC
  __builtin_amdgcn_global_load_async_to_lds_b128(
      (swin_g128p)(void*)g, (swin_l128p)(void*)l, 0, 0);
#else
  *(v8h*)l = *(const v8h*)g;
#endif
}
__device__ __forceinline__ void swin_wait_async_keep3() {
#if SWIN_ASYNC
  __builtin_amdgcn_s_wait_asynccnt(3);   // one 3-instr group still in flight
#endif
}
__device__ __forceinline__ void swin_wait_async_all() {
#if SWIN_ASYNC
  __builtin_amdgcn_s_wait_asynccnt(0);
#endif
}

__device__ __forceinline__ float swin_gelu(float v) {
  return 0.5f * v * (1.0f + erff(v * 0.7071067811865475f));
}

// WMMA 16-bit operand fragment: lane-half hs covers K {8hs..8hs+7, 16+8hs..16+8hs+7}
// -> two contiguous 16B LDS loads merged into one v16h.
__device__ __forceinline__ v16h swin_frag(const h16* rowPtr, int hs) {
  v8h lo = *(const v8h*)(rowPtr + 8 * hs);
  v8h hi = *(const v8h*)(rowPtr + 16 + 8 * hs);
  return __builtin_shufflevector(lo, hi, 0, 1, 2, 3, 4, 5, 6, 7,
                                 8, 9, 10, 11, 12, 13, 14, 15);
}

// ---------------------------------------------------------------------------
// Weight cast+transpose: W[K,N] f32 -> Wt[N,K] f16 (contiguous writes)
// ---------------------------------------------------------------------------
__global__ void swin_wcast(const float* __restrict__ W, h16* __restrict__ Wt,
                           int K, int N) {
  size_t i = (size_t)blockIdx.x * blockDim.x + threadIdx.x;
  size_t total = (size_t)K * N;
  if (i >= total) return;
  int n = (int)(i / K);
  int k = (int)(i % K);
  Wt[i] = (h16)W[(size_t)k * N + n];
}

// ---------------------------------------------------------------------------
// Patch embed: 4x4/s4 conv on NCHW input -> (B,56,56,96) token layout (f32)
// ---------------------------------------------------------------------------
__global__ void swin_patch_embed(const float* __restrict__ x,
                                 const float* __restrict__ w,
                                 const float* __restrict__ bias,
                                 float* __restrict__ out,
                                 int B, int H, int W, int C) {
  size_t i = (size_t)blockIdx.x * blockDim.x + threadIdx.x;
  size_t total = (size_t)B * H * W * C;
  if (i >= total) return;
  int oc = (int)(i % C);
  size_t t = i / C;
  int ow = (int)(t % W);
  int oh = (int)((t / W) % H);
  int bb = (int)(t / ((size_t)W * H));
  int IH = H * 4, IW = W * 4;
  float acc = bias[oc];
  for (int ic = 0; ic < 3; ++ic)
    for (int kh = 0; kh < 4; ++kh)
      for (int kw = 0; kw < 4; ++kw)
        acc += x[(((size_t)bb * 3 + ic) * IH + oh * 4 + kh) * IW + ow * 4 + kw] *
               w[((oc * 3 + ic) * 4 + kh) * 4 + kw];
  out[i] = acc;
}

// ---------------------------------------------------------------------------
// LayerNorm: one wave per token, shuffle reduction (wave32).
// out_half=1 -> write f16 (GEMM A-operand), else f32.
// ---------------------------------------------------------------------------
__global__ void swin_ln(const float* __restrict__ src, void* __restrict__ dst,
                        const float* __restrict__ g, const float* __restrict__ b,
                        int N, int C, int out_half) {
  int wpb  = blockDim.x >> 5;
  int wid  = threadIdx.x >> 5;
  int lane = threadIdx.x & 31;
  int tok  = blockIdx.x * wpb + wid;
  if (tok >= N) return;
  const float* row = src + (size_t)tok * C;
  float buf[48];                 // C <= 1536 -> <= 48 per lane
  int nv = C >> 5;
  float s0 = 0.f, s1 = 0.f;
  for (int i = 0; i < nv; ++i) {
    float v = row[i * 32 + lane];
    buf[i] = v; s0 += v; s1 += v * v;
  }
  for (int off = 16; off > 0; off >>= 1) {
    s0 += __shfl_xor(s0, off, 32);
    s1 += __shfl_xor(s1, off, 32);
  }
  float mean = s0 / C;
  float var  = s1 / C - mean * mean;
  float rstd = rsqrtf(var + 1e-5f);
  if (out_half) {
    h16* drow = (h16*)dst + (size_t)tok * C;
    for (int i = 0; i < nv; ++i) {
      int c = i * 32 + lane;
      drow[c] = (h16)((buf[i] - mean) * rstd * g[c] + b[c]);
    }
  } else {
    float* drow = (float*)dst + (size_t)tok * C;
    for (int i = 0; i < nv; ++i) {
      int c = i * 32 + lane;
      drow[c] = (buf[i] - mean) * rstd * g[c] + b[c];
    }
  }
}

// ---------------------------------------------------------------------------
// WMMA GEMM: D[M,N] = A[M,K](f16) * Wt[N,K](f16, pre-transposed) (+bias)
//            (+GELU | +residual).  128x64 tile, 8 waves, K-step 32,
// double-buffered LDS fed by async global->LDS b128 copies (ASYNCcnt).
// No fill-side bounds guards: OOB rows/cols only produce discarded D elements
// (all source buffers carry pad slack), keeping EXEC uniform for async issue.
// ---------------------------------------------------------------------------
__global__ __launch_bounds__(256) void swin_gemm(
    const h16* __restrict__ A, const h16* __restrict__ Wt,
    const float* __restrict__ bias, const float* __restrict__ res,
    void* __restrict__ Dv, int M, int N, int K, int epi, int out_half) {
  __shared__ h16 As[2][128][40];   // [buf][m][k], 80B pitch
  __shared__ h16 Bs[2][64][40];    // [buf][n][k], 80B pitch
  int tid  = threadIdx.x;
  int wave = tid >> 5, lane = tid & 31;
  int r = lane & 15, hs = lane >> 4;
  int mBase = blockIdx.y * 128, nBase = blockIdx.x * 64;
  const h16* Ag = A  + (size_t)mBase * K;
  const h16* Bg = Wt + (size_t)nBase * K;

  // Per wave per group: 2 async instr (A) + 1 (B) = 3.
  auto issue = [&](int buf, int kb) {
    for (int j = 0; j < 2; ++j) {
      int i  = tid + j * 256;          // 512 chunks: 128 rows x 4 chunks
      int rr = i >> 2, qc = (i & 3) * 8;
      swin_cp16(Ag + (size_t)rr * K + kb + qc, &As[buf][rr][qc]);
    }
    {
      int nn = tid >> 2, kc = (tid & 3) * 8;  // 256 chunks: 64 rows x 4
      swin_cp16(Bg + (size_t)nn * K + kb + kc, &Bs[buf][nn][kc]);
    }
  };

  v8f acc[4];
  for (int i = 0; i < 4; ++i)
    for (int j = 0; j < 8; ++j) acc[i][j] = 0.f;

  issue(0, 0);
  if (32 < K) issue(1, 32);

  int ib = 0;
  for (int kb = 0; kb < K; kb += 32, ib ^= 1) {
    if (kb + 32 < K) swin_wait_async_keep3();
    else             swin_wait_async_all();
    __syncthreads();
    int mRow = wave * 16 + r;
    v16h a = swin_frag(&As[ib][mRow][0], hs);
    for (int ct = 0; ct < 4; ++ct) {
      v16h bf = swin_frag(&Bs[ib][ct * 16 + r][0], hs);
      acc[ct] = __builtin_amdgcn_wmma_f32_16x16x32_f16(
          false, a, false, bf, (short)0, acc[ct], false, false);
    }
    __syncthreads();
    if (kb + 64 < K) issue(ib, kb + 64);
  }

  float* Df = (float*)Dv;
  h16*   Dh = (h16*)Dv;
  for (int ct = 0; ct < 4; ++ct) {
    int n = nBase + ct * 16 + r;
    if (n >= N) continue;
    float bv = bias ? bias[n] : 0.f;
    for (int i = 0; i < 8; ++i) {
      int m = mBase + wave * 16 + hs * 8 + i;
      if (m >= M) continue;
      float v = acc[ct][i] + bv;
      if (epi == EPI_BIAS_GELU)      v = swin_gelu(v);
      else if (epi == EPI_BIAS_RES)  v += res[(size_t)m * N + n];
      size_t o = (size_t)m * N + n;
      if (out_half) Dh[o] = (h16)v; else Df[o] = v;
    }
  }
}

// ---------------------------------------------------------------------------
// Window attention: one workgroup per (batch, window, head).
// qkv input and output are f16; shift/partition/reverse folded into indexing.
// QK^T and PV via WMMA with 16B vectorized fragment loads.
// ---------------------------------------------------------------------------
__global__ __launch_bounds__(128) void swin_attn(
    const h16* __restrict__ qkv, const float* __restrict__ rel,
    h16* __restrict__ out, int H, int W, int C, int nh, int w1, int shifted) {
  __shared__ h16 Qs[64][40], Ks[64][40];   // [token][ch]
  __shared__ h16 Vt[32][72];               // [ch][token]
  __shared__ float Ss[64][68];
  __shared__ h16 Ps[64][72];               // [qtoken][ktoken]

  int win = blockIdx.x, head = blockIdx.y, b = blockIdx.z;
  int wi = win / w1, wj = win % w1;
  int sh = shifted ? 3 : 0;
  int C3 = 3 * C;
  int qo = head * 32, ko = (nh + head) * 32, vo = (2 * nh + head) * 32;
  int tid = threadIdx.x;

  for (int i = tid; i < 512; i += 128) {   // 64 tokens x 32 ch / 4
    int row = i >> 3, c4 = (i & 7) * 4;
    v4h q = {}, k = {}, v = {};
    if (row < 49) {
      int pi = row / 7, pj = row % 7;
      int hh = (wi * 7 + pi + sh) % H;
      int ww = (wj * 7 + pj + sh) % W;
      size_t t = ((size_t)b * H + hh) * W + ww;
      const h16* base = qkv + t * C3;
      q = *(const v4h*)(base + qo + c4);
      k = *(const v4h*)(base + ko + c4);
      v = *(const v4h*)(base + vo + c4);
    }
    *(v4h*)&Qs[row][c4] = q;
    *(v4h*)&Ks[row][c4] = k;
    Vt[c4 + 0][row] = v[0];
    Vt[c4 + 1][row] = v[1];
    Vt[c4 + 2][row] = v[2];
    Vt[c4 + 3][row] = v[3];
  }
  __syncthreads();

  int wave = tid >> 5, lane = tid & 31, r = lane & 15, hs = lane >> 4;
  int mRow = wave * 16 + r;
  const float scale = 0.17677669529663687f;  // 1/sqrt(32)

  {
    v16h a = swin_frag(&Qs[mRow][0], hs);
    for (int ct = 0; ct < 4; ++ct) {
      v16h bf = swin_frag(&Ks[ct * 16 + r][0], hs);  // B = K^T: key-major rows
      v8f s;
      for (int j = 0; j < 8; ++j) s[j] = 0.f;
      s = __builtin_amdgcn_wmma_f32_16x16x32_f16(
          false, a, false, bf, (short)0, s, false, false);
      for (int i = 0; i < 8; ++i)
        Ss[wave * 16 + hs * 8 + i][ct * 16 + r] = s[i] * scale;
    }
  }
  __syncthreads();

  const float* relh = rel + head * 13 * 13;
  if (tid < 64) {
    int i = tid;
    if (i < 49) {
      int pi = i / 7, pj = i % 7;
      bool wiL = (wi == w1 - 1), wjL = (wj == w1 - 1);
      float mx = -3.0e38f;
      for (int j = 0; j < 64; ++j) {
        float v = -3.0e38f;
        if (j < 49) {
          int ki = j / 7, kj = j % 7;
          v = Ss[i][j] + relh[(pi - ki + 6) * 13 + (pj - kj + 6)];
          if (shifted) {
            bool m1 = wiL && ((pi >= 4) != (ki >= 4));
            bool m2 = wjL && ((pj >= 4) != (kj >= 4));
            if (m1 || m2) v = -3.0e38f;
          }
        }
        Ss[i][j] = v;
        if (v > mx) mx = v;
      }
      float sum = 0.f;
      for (int j = 0; j < 64; ++j) {
        float v = Ss[i][j];
        float e = (v <= -1.0e38f) ? 0.f : expf(v - mx);
        Ps[i][j] = (h16)e;
        sum += e;
      }
      float inv = 1.f / sum;
      for (int j = 0; j < 64; ++j)
        Ps[i][j] = (h16)((float)Ps[i][j] * inv);
    } else {
      for (int j = 0; j < 64; ++j) Ps[i][j] = (h16)0.f;
    }
  }
  __syncthreads();

  v8f o[2];
  for (int c2 = 0; c2 < 2; ++c2)
    for (int j = 0; j < 8; ++j) o[c2][j] = 0.f;
  for (int kb = 0; kb < 64; kb += 32) {
    v16h pa = swin_frag(&Ps[mRow][kb], hs);
    for (int ct = 0; ct < 2; ++ct) {
      v16h bf = swin_frag(&Vt[ct * 16 + r][kb], hs);  // channel-major rows
      o[ct] = __builtin_amdgcn_wmma_f32_16x16x32_f16(
          false, pa, false, bf, (short)0, o[ct], false, false);
    }
  }
  for (int ct = 0; ct < 2; ++ct) {
    for (int i = 0; i < 8; ++i) {
      int m = wave * 16 + hs * 8 + i;
      if (m < 49) {
        int pi = m / 7, pj = m % 7;
        int hh = (wi * 7 + pi + sh) % H;
        int ww = (wj * 7 + pj + sh) % W;
        size_t t = ((size_t)b * H + hh) * W + ww;
        out[t * C + head * 32 + ct * 16 + r] = (h16)o[ct][i];
      }
    }
  }
}

// ---------------------------------------------------------------------------
// Patch merging gather: (B,H,W,C) -> (B,H/2,W/2,4C), channel = (w2*2+h2)*C+c
// ---------------------------------------------------------------------------
__global__ void swin_merge_gather(const float* __restrict__ x,
                                  float* __restrict__ out,
                                  int B, int H, int W, int C) {
  int H2 = H / 2, W2 = W / 2, C4 = 4 * C;
  size_t i = (size_t)blockIdx.x * blockDim.x + threadIdx.x;
  size_t total = (size_t)B * H2 * W2 * C4;
  if (i >= total) return;
  int cn = (int)(i % C4);
  size_t t = i / C4;
  int ww = (int)(t % W2);
  int hh = (int)((t / W2) % H2);
  int b  = (int)(t / ((size_t)W2 * H2));
  int w2 = cn / (2 * C);
  int rem = cn % (2 * C);
  int h2 = rem / C;
  int c  = rem % C;
  out[i] = x[(((size_t)b * H + 2 * hh + h2) * W + 2 * ww + w2) * C + c];
}

// ---------------------------------------------------------------------------
// Host orchestration
// ---------------------------------------------------------------------------
extern "C" void kernel_launch(void* const* d_in, const int* in_sizes, int n_in,
                              void* d_out, int out_size, void* d_ws, size_t ws_size,
                              hipStream_t stream) {
  (void)in_sizes; (void)n_in; (void)out_size; (void)ws_size;
  int idx = 0;
  auto in = [&]() -> const float* { return (const float*)d_in[idx++]; };

  const float* X      = in();
  const float* conv_w = in();
  const float* conv_b = in();
  const float* ln0_g  = in();
  const float* ln0_b  = in();

  struct Blk {
    const float *ln1_g, *ln1_b, *qkv_w, *qkv_b, *rel, *proj_w, *proj_b,
                *ln2_g, *ln2_b, *mlp_w1, *mlp_b1, *mlp_w2, *mlp_b2;
  };
  auto rdBlk = [&]() {
    Blk b;
    b.ln1_g = in();  b.ln1_b = in();
    b.qkv_w = in();  b.qkv_b = in();  b.rel = in();
    b.proj_w = in(); b.proj_b = in();
    b.ln2_g = in();  b.ln2_b = in();
    b.mlp_w1 = in(); b.mlp_b1 = in();
    b.mlp_w2 = in(); b.mlp_b2 = in();
    return b;
  };

  Blk s1[2]; for (int i = 0; i < 2; ++i) s1[i] = rdBlk();
  const float* m2_g = in(); const float* m2_b = in(); const float* m2_w = in();
  Blk s2[2]; for (int i = 0; i < 2; ++i) s2[i] = rdBlk();
  const float* m3_g = in(); const float* m3_b = in(); const float* m3_w = in();
  Blk s3[6]; for (int i = 0; i < 6; ++i) s3[i] = rdBlk();
  const float* m4_g = in(); const float* m4_b = in(); const float* m4_w = in();
  Blk s4[2]; for (int i = 0; i < 2; ++i) s4[i] = rdBlk();

  // Stage outputs (residual stream, f32) live directly in d_out.
  float* out_f = (float*)d_out;
  float* R0 = out_f;
  float* R1 = R0 + (size_t)16 * 56 * 56 * 96;
  float* R2 = R1 + (size_t)16 * 28 * 28 * 192;
  float* R3 = R2 + (size_t)16 * 14 * 14 * 384;

  float* ws   = (float*)d_ws;
  h16*   wLNh = (h16*)ws;                   // LN / merged-LN out, <= 4.82M halves
  h16*   wATT = (h16*)(ws + 2500000);       // attention out, <= 4.82M halves
  h16*   wBIG = (h16*)(ws + 5000000);       // qkv / mlp hidden, <= 19.3M halves
  float* wMG  = ws + 15000000;              // merge gather f32, <= 4.82M floats
  h16*   wWT  = (h16*)(ws + 20000000);      // f16 W^T, <= 2.36M halves

  auto gemm = [&](const h16* A, const float* Wm, const float* bias,
                  const float* res, void* D, int M, int N, int K, int epi,
                  int out_half) {
    size_t wn = (size_t)K * N;
    swin_wcast<<<(unsigned)((wn + 255) / 256), 256, 0, stream>>>(Wm, wWT, K, N);
    dim3 g((N + 63) / 64, (M + 127) / 128);
    swin_gemm<<<g, 256, 0, stream>>>(A, wWT, bias, res, D, M, N, K, epi, out_half);
  };
  auto ln = [&](const float* src, void* dst, const float* g, const float* b,
                int N, int C, int out_half) {
    int wpb = 8;
    dim3 gr((N + wpb - 1) / wpb);
    swin_ln<<<gr, wpb * 32, 0, stream>>>(src, dst, g, b, N, C, out_half);
  };
  auto stage = [&](float* x, int H, int W, int C, const Blk* blks, int nb,
                   bool shift_ok) {
    int N = 16 * H * W, nh = C / 32, w1 = H / 7;
    for (int i = 0; i < nb; ++i) {
      const Blk& bp = blks[i];
      int shifted = (shift_ok && (i & 1)) ? 1 : 0;
      ln(x, wLNh, bp.ln1_g, bp.ln1_b, N, C, 1);
      gemm(wLNh, bp.qkv_w, bp.qkv_b, nullptr, wBIG, N, 3 * C, C, EPI_BIAS, 1);
      dim3 ga(w1 * w1, nh, 16);
      swin_attn<<<ga, 128, 0, stream>>>(wBIG, bp.rel, wATT, H, W, C, nh, w1, shifted);
      gemm(wATT, bp.proj_w, bp.proj_b, x, x, N, C, C, EPI_BIAS_RES, 0);
      ln(x, wLNh, bp.ln2_g, bp.ln2_b, N, C, 1);
      gemm(wLNh, bp.mlp_w1, bp.mlp_b1, nullptr, wBIG, N, 4 * C, C, EPI_BIAS_GELU, 1);
      gemm(wBIG, bp.mlp_w2, bp.mlp_b2, x, x, N, C, 4 * C, EPI_BIAS_RES, 0);
    }
  };
  auto merge = [&](const float* xin, int H, int W, int C, const float* g,
                   const float* b, const float* red, float* xout) {
    int H2 = H / 2, W2 = W / 2, C4 = 4 * C;
    size_t tot = (size_t)16 * H2 * W2 * C4;
    swin_merge_gather<<<(unsigned)((tot + 255) / 256), 256, 0, stream>>>(
        xin, wMG, 16, H, W, C);
    ln(wMG, wLNh, g, b, 16 * H2 * W2, C4, 1);
    gemm(wLNh, red, nullptr, nullptr, xout, 16 * H2 * W2, 2 * C, C4, EPI_BIAS, 0);
  };

  // Patch embed + LN0 -> stage1 input (in d_out region 0, f32 in-place)
  {
    size_t tot = (size_t)16 * 56 * 56 * 96;
    swin_patch_embed<<<(unsigned)((tot + 255) / 256), 256, 0, stream>>>(
        X, conv_w, conv_b, R0, 16, 56, 56, 96);
    ln(R0, R0, ln0_g, ln0_b, 16 * 56 * 56, 96, 0);
  }

  stage(R0, 56, 56, 96,  s1, 2, true);
  merge(R0, 56, 56, 96,  m2_g, m2_b, m2_w, R1);
  stage(R1, 28, 28, 192, s2, 2, true);
  merge(R1, 28, 28, 192, m3_g, m3_b, m3_w, R2);
  stage(R2, 14, 14, 384, s3, 6, true);
  merge(R2, 14, 14, 384, m4_g, m4_b, m4_w, R3);
  stage(R3, 7, 7, 768,   s4, 2, false);
}
